// SequenceMemory_71588514889995
// MI455X (gfx1250) — matrix operand order repaired and checked by reading, not compiled, not run on hardware
//
#include <hip/hip_runtime.h>
#include <hip/hip_bf16.h>

#define S_LEN 1024
#define NTS   10
#define NN    2048
#define CA3N  2048
#define SPARS 0.05f
#define LRV   0.01f

typedef __attribute__((ext_vector_type(16))) _Float16 v16h;
typedef __attribute__((ext_vector_type(8)))  float    v8f;
typedef __attribute__((ext_vector_type(2)))  float    v2f;

union Frag16 { v16h v; uint4 u[2]; };

__device__ __forceinline__ unsigned wave_ballot(bool p) {
#if __has_builtin(__builtin_amdgcn_ballot_w32)
    return __builtin_amdgcn_ballot_w32(p);
#else
    return (unsigned)__ballot(p);
#endif
}

// CDNA5 async global->LDS copy (ASYNCcnt-tracked, bypasses VGPRs).
__device__ __forceinline__ void async_copy_b128(void* lds_dst, const void* gsrc) {
    unsigned           loff = (unsigned)(size_t)lds_dst;
    unsigned long long gptr = (unsigned long long)(size_t)gsrc;
    asm volatile("global_load_async_to_lds_b128 %0, %1, off"
                 :: "v"(loff), "v"(gptr) : "memory");
}
__device__ __forceinline__ void wait_async0() {
    asm volatile("s_wait_asynccnt 0x0" ::: "memory");
}

// Forced VOP3P packed fp32 math (clang scalarizes <2 x float> otherwise).
__device__ __forceinline__ v2f pk_fma(v2f a, v2f b, v2f c) {
    v2f d;
    asm("v_pk_fma_f32 %0, %1, %2, %3" : "=v"(d) : "v"(a), "v"(b), "v"(c));
    return d;
}
__device__ __forceinline__ v2f pk_mul(v2f a, v2f b) {
    v2f d;
    asm("v_pk_mul_f32 %0, %1, %2" : "=v"(d) : "v"(a), "v"(b));
    return d;
}
__device__ __forceinline__ v2f pk_add(v2f a, v2f b) {
    v2f d;
    asm("v_pk_add_f32 %0, %1, %2" : "=v"(d) : "v"(a), "v"(b));
    return d;
}

// ---------------------------------------------------------------------------
// Phase 1: spike encode.  hippo16[s,n] = sum_t tok_spike * (1 + 0.5*pos_spike)
// Values are multiples of 0.5 in [0,15] -> exact in f16.
// ---------------------------------------------------------------------------
__global__ void k_encode(const int* __restrict__ tok,
                         const float* __restrict__ sdr,
                         const float* __restrict__ pos,
                         _Float16* __restrict__ hippo16) {
    int id = blockIdx.x * blockDim.x + threadIdx.x;   // S_LEN*NN threads exact
    int s = id >> 11;
    int n = id & (NN - 1);
    int t0 = tok[s];
    size_t sb = (size_t)t0 * NTS * NN + n;
    size_t pb = (size_t)s  * NTS * NN + n;
    float acc = 0.0f;
#pragma unroll
    for (int t = 0; t < NTS; ++t) {
        float ts = sdr[sb + (size_t)t * NN];
        float ps = pos[pb + (size_t)t * NN];
        float tv = (ts < SPARS) ? 1.0f : 0.0f;
        float pv = (ps < SPARS) ? 0.5f : 0.0f;
        acc += tv * (1.0f + pv);
    }
    hippo16[id] = (_Float16)acc;
}

// ---------------------------------------------------------------------------
// Phase 2a: convert w_hippo fp32 -> f16 (row-major [CA3, N], read along N=K)
// ---------------------------------------------------------------------------
__global__ void k_cvt(const float* __restrict__ w, _Float16* __restrict__ o, int n) {
    int id = blockIdx.x * blockDim.x + threadIdx.x;
    if (id < n) o[id] = (_Float16)w[id];
}

// ---------------------------------------------------------------------------
// Phase 2b: WMMA GEMM  C[s,c] = sum_k hippo16[s,k] * wh16[c,k], threshold > 0,
// packed to bitmask bits[s][c/32].  Block tile 128x128, 8 waves, wave = 16 M
// rows x 128 N cols (8 accumulators).  K staged via async-to-LDS, 32/iter.
// ---------------------------------------------------------------------------
__global__ __launch_bounds__(256) void k_gemm(const _Float16* __restrict__ A,
                                              const _Float16* __restrict__ B,
                                              unsigned* __restrict__ bits) {
    __shared__ _Float16 ldsA[128 * 32];
    __shared__ _Float16 ldsB[128 * 32];

    const int tid  = threadIdx.x;
    const int lane = tid & 31;
    const int wid  = tid >> 5;
    const int bm   = blockIdx.x * 128;
    const int bn   = blockIdx.y * 128;

    v8f acc[8];
#pragma unroll
    for (int i = 0; i < 8; ++i) acc[i] = (v8f){0, 0, 0, 0, 0, 0, 0, 0};

    // cooperative staging: thread -> (row, 16-half chunk), 16 B per thread
    const int lrow  = tid >> 1;
    const int lpart = (tid & 1) * 16;
    const _Float16* gA = A + (size_t)(bm + lrow) * NN + lpart;
    const _Float16* gB = B + (size_t)(bn + lrow) * NN + lpart;
    _Float16* dstA = &ldsA[lrow * 32 + lpart];
    _Float16* dstB = &ldsB[lrow * 32 + lpart];

    const int m  = lane & 15;
    const int hi = lane >> 4;

    for (int kt = 0; kt < NN; kt += 32) {
        __syncthreads();                    // prior ds_loads done before overwrite
        async_copy_b128(dstA, gA + kt);
        async_copy_b128(dstB, gB + kt);
        if (kt + 32 < NN) {                 // pull next K tiles toward L2
            __builtin_prefetch(gA + kt + 32, 0, 1);
            __builtin_prefetch(gB + kt + 32, 0, 1);
        }
        wait_async0();
        __syncthreads();

        // A fragment 16x32 f16: lanes 0-15: VGPR0-3=K0..7, VGPR4-7=K16..23;
        // lanes 16-31: K8..15 / K24..31.
        Frag16 fa, fb[8];
        const _Float16* ap = &ldsA[(wid * 16 + m) * 32];
        fa.u[0] = *(const uint4*)(ap + 8 * hi);
        fa.u[1] = *(const uint4*)(ap + 16 + 8 * hi);
#pragma unroll
        for (int nt = 0; nt < 8; ++nt) {
            // B fragment 32x16 f16: lane = N col, lanes 0-15: K0..7/K16..23, lanes 16-31: K8..15/K24..31
            const _Float16* bp = &ldsB[(nt * 16 + m) * 32];
            fb[nt].u[0] = *(const uint4*)(bp + 8 * hi);
            fb[nt].u[1] = *(const uint4*)(bp + 16 + 8 * hi);
        }
#pragma unroll
        for (int nt = 0; nt < 8; ++nt) {
            acc[nt] = __builtin_amdgcn_wmma_f32_16x16x32_f16(
                false, fa.v, false, fb[nt].v, (short)0, acc[nt], false, false);
        }
#if __has_builtin(__builtin_amdgcn_sched_group_barrier)
        // schedule: all 18 DS reads first, then the 8 WMMAs back-to-back
        __builtin_amdgcn_sched_group_barrier(0x100, 18, 0);  // DS read
        __builtin_amdgcn_sched_group_barrier(0x008, 8, 0);   // MFMA/WMMA
#endif
    }

    // Threshold + bitpack via ballot: C layout -> ballot bits 0-15 = row v,
    // bits 16-31 = row v+8 (cols 0-15 of the 16x16 tile).  Single writer.
#pragma unroll
    for (int p = 0; p < 4; ++p) {
        unsigned b0[8], b1[8];
#pragma unroll
        for (int v = 0; v < 8; ++v) {
            b0[v] = wave_ballot(acc[2 * p][v] > 0.0f);
            b1[v] = wave_ballot(acc[2 * p + 1][v] > 0.0f);
        }
        if (lane == 0) {
#pragma unroll
            for (int v = 0; v < 8; ++v) {
                unsigned lo = (b0[v] & 0xffffu) | ((b1[v] & 0xffffu) << 16);
                unsigned hw = (b0[v] >> 16) | (b1[v] & 0xffff0000u);
                int r0 = bm + wid * 16 + v;
                bits[(size_t)r0 * 64 + (bn >> 5) + p]       = lo;
                bits[(size_t)(r0 + 8) * 64 + (bn >> 5) + p] = hw;
            }
        }
    }
}

// ---------------------------------------------------------------------------
// Phase 3: sequential Hebbian scan, one wave per W row.  Row (2048 f32) lives
// in 32 float2 VGPR pairs/lane; packed VOP3P f32 math forced via inline asm.
// Per step: masked packed add of LR, packed FMA sum-of-squares, 5x shfl_xor
// wave reduction, scale by 1/max(||row||,1).  No barriers, no LDS.
// ---------------------------------------------------------------------------
__global__ __launch_bounds__(256) void k_scan(const float* __restrict__ w0,
                                              const unsigned* __restrict__ bits,
                                              float* __restrict__ out) {
    const int lane = threadIdx.x & 31;
    const int wid  = threadIdx.x >> 5;
    const int row  = blockIdx.x * 8 + wid;

    v2f w[32];                              // 64 contiguous cols: lane*64 ..
    const float* src = w0 + (size_t)row * CA3N + lane * 64;
#pragma unroll
    for (int j4 = 0; j4 < 16; ++j4) {
        float4 v = *(const float4*)(src + j4 * 4);
        w[j4 * 2 + 0] = (v2f){v.x, v.y};
        w[j4 * 2 + 1] = (v2f){v.z, v.w};
    }

    const int pwidx = row >> 5;
    const int pbit  = row & 31;

    for (int t = 0; t < S_LEN - 1; ++t) {
        const unsigned* bt = bits + (size_t)t * 64;
        unsigned pw   = bt[64 + pwidx];                     // post = ca3[t+1][row]
        float    addv = ((pw >> pbit) & 1u) ? LRV : 0.0f;
        unsigned p0 = bt[2 * lane];                         // pre bits, cols lane*64..
        unsigned p1 = bt[2 * lane + 1];
#pragma unroll
        for (int j = 0; j < 16; ++j) {
            v2f a = (v2f){((p0 >> (2 * j)) & 1u) ? addv : 0.0f,
                          ((p0 >> (2 * j + 1)) & 1u) ? addv : 0.0f};
            w[j] = pk_add(w[j], a);
        }
#pragma unroll
        for (int j = 0; j < 16; ++j) {
            v2f a = (v2f){((p1 >> (2 * j)) & 1u) ? addv : 0.0f,
                          ((p1 >> (2 * j + 1)) & 1u) ? addv : 0.0f};
            w[16 + j] = pk_add(w[16 + j], a);
        }

        v2f q0 = (v2f){0, 0}, q1 = q0, q2 = q0, q3 = q0;
#pragma unroll
        for (int j = 0; j < 32; j += 4) {
            q0 = pk_fma(w[j + 0], w[j + 0], q0);
            q1 = pk_fma(w[j + 1], w[j + 1], q1);
            q2 = pk_fma(w[j + 2], w[j + 2], q2);
            q3 = pk_fma(w[j + 3], w[j + 3], q3);
        }
        v2f qs = pk_add(pk_add(q0, q1), pk_add(q2, q3));
        float s = qs.x + qs.y;
#pragma unroll
        for (int off = 16; off > 0; off >>= 1) s += __shfl_xor(s, off, 32);

        float scale = 1.0f / fmaxf(sqrtf(s), 1.0f);
        v2f sc = (v2f){scale, scale};
#pragma unroll
        for (int j = 0; j < 32; ++j) w[j] = pk_mul(w[j], sc);
    }

    float* dst = out + (size_t)row * CA3N + lane * 64;
#pragma unroll
    for (int j4 = 0; j4 < 16; ++j4) {
        *(float4*)(dst + j4 * 4) = make_float4(w[j4 * 2 + 0].x, w[j4 * 2 + 0].y,
                                               w[j4 * 2 + 1].x, w[j4 * 2 + 1].y);
    }
}

// ---------------------------------------------------------------------------
extern "C" void kernel_launch(void* const* d_in, const int* in_sizes, int n_in,
                              void* d_out, int out_size, void* d_ws, size_t ws_size,
                              hipStream_t stream) {
    (void)in_sizes; (void)n_in; (void)out_size; (void)ws_size;
    const int*   tok    = (const int*)d_in[0];
    const float* sdr    = (const float*)d_in[1];
    const float* pos    = (const float*)d_in[2];
    const float* whippo = (const float*)d_in[3];
    const float* assoc  = (const float*)d_in[4];
    float*       out    = (float*)d_out;

    char* ws = (char*)d_ws;
    _Float16* hippo16 = (_Float16*)ws;                                   // 4 MB
    _Float16* wh16    = (_Float16*)(ws + (size_t)4 * 1024 * 1024);       // 8 MB
    unsigned* bits    = (unsigned*)(ws + (size_t)12 * 1024 * 1024);      // 256 KB

    k_encode<<<(S_LEN * NN) / 256, 256, 0, stream>>>(tok, sdr, pos, hippo16);
    k_cvt<<<(CA3N * NN) / 256, 256, 0, stream>>>(whippo, wh16, CA3N * NN);
    dim3 g3(S_LEN / 128, CA3N / 128);
    k_gemm<<<g3, 256, 0, stream>>>(hippo16, wh16, bits);
    k_scan<<<CA3N / 8, 256, 0, stream>>>(assoc, bits, out);
}
